// Quantize_35338990911834
// MI455X (gfx1250) — matrix operand order repaired and checked by reading, not compiled, not run on hardware
//
#include <hip/hip_runtime.h>
#include <hip/hip_bf16.h>

typedef __bf16 bf16_t;
typedef __bf16 bf16x8  __attribute__((ext_vector_type(8)));
typedef __bf16 bf16x16 __attribute__((ext_vector_type(16)));
typedef float  f32x8   __attribute__((ext_vector_type(8)));

#define DIM    256
#define NCODE  1024
#define HW     4096
#define NPIX   32768           // 8 * 64 * 64
#define NOUT_Z 8388608         // 8 * 256 * 64 * 64
#define LDS_STRIDE 136         // 128 ch + 8 pad (16B aligned rows, low bank conflict)

__device__ __forceinline__ unsigned f32_to_bf16_rne(float f) {
    unsigned u = __float_as_uint(f);
    return (u + 0x7FFFu + ((u >> 16) & 1u)) >> 16;
}

// ---------------- Kernel 0: codebook -> bf16, plus ||e||^2 ----------------
__global__ __launch_bounds__(256) void prep_codebook(
    const float* __restrict__ cb, unsigned short* __restrict__ cbb,
    float* __restrict__ e2) {
    __shared__ float red[256];
    const int j = blockIdx.x;            // code id
    const int c = threadIdx.x;           // channel
    float v = cb[j * DIM + c];
    cbb[j * DIM + c] = (unsigned short)f32_to_bf16_rne(v);
    red[c] = v * v;
    __syncthreads();
    for (int s = 128; s > 0; s >>= 1) {
        if (c < s) red[c] += red[c + s];
        __syncthreads();
    }
    if (c == 0) e2[j] = red[0];
}

// ---------------- Kernel 1: WMMA distance + argmin ----------------
// Block: 128 threads = 4 wave32, 128 pixels per block.
// Wave w owns 2 pixel tiles (32 pixels): rows [w*32, w*32+16) and [w*32+16, w*32+32).
// A (z tile, bf16) staged via LDS in two 128-channel phases, then held full-K in VGPRs
// (2 x 64 VGPRs). B streamed once per wave from the L2-resident bf16 codebook and shared
// by both M tiles -> 2 WMMAs per B-load pair. score = ||e||^2 - 2*(z.e); argmin over 1024.
__global__ __launch_bounds__(128) void argmin_kernel(
    const float* __restrict__ z, const bf16_t* __restrict__ cbb,
    const float* __restrict__ e2, int* __restrict__ idxout) {
    __shared__ __attribute__((aligned(16))) bf16_t zs[128 * LDS_STRIDE];

    const int tid = threadIdx.x;
    const int n0  = blockIdx.x * 128;          // first pixel of this block
    const int b   = n0 >> 12;                  // batch (4096 % 128 == 0 -> no crossing)
    const int hw0 = n0 & 4095;
    const float* zb = z + (size_t)b * DIM * HW + hw0;

    const int lane = tid & 31;
    const int wave = tid >> 5;                 // 0..3
    const int m    = lane & 15;
    const int hi   = lane >> 4;                // lane half
    const int row0 = wave * 32 + m;            // M tile 0
    const int row1 = row0 + 16;                // M tile 1

    bf16x16 a0[8], a1[8];

    // two-phase staging: channels [phase*128, phase*128+128)
    #pragma unroll
    for (int phase = 0; phase < 2; ++phase) {
        {
            const int p = tid;                 // pixel within tile (128 threads)
            #pragma unroll 4
            for (int cl = 0; cl < 128; cl += 2) {
                const int c = phase * 128 + cl;
                float f0 = zb[(size_t)c       * HW + p];
                float f1 = zb[(size_t)(c + 1) * HW + p];
                unsigned packed = f32_to_bf16_rne(f0) | (f32_to_bf16_rne(f1) << 16);
                ((unsigned*)zs)[(p * LDS_STRIDE + cl) >> 1] = packed;
            }
        }
        __syncthreads();
        // A fragment layout (16-bit A 16x32): lanes 0-15: K 0-7 & 16-23; lanes 16-31: K 8-15 & 24-31
        {
            const bf16_t* ap0 = &zs[row0 * LDS_STRIDE];
            const bf16_t* ap1 = &zs[row1 * LDS_STRIDE];
            const int koff = hi * 8;
            #pragma unroll
            for (int kc = 0; kc < 4; ++kc) {
                bf16x8 lo0 = *(const bf16x8*)(ap0 + kc * 32 + koff);
                bf16x8 hh0 = *(const bf16x8*)(ap0 + kc * 32 + koff + 16);
                a0[phase * 4 + kc] = __builtin_shufflevector(
                    lo0, hh0, 0,1,2,3,4,5,6,7,8,9,10,11,12,13,14,15);
                bf16x8 lo1 = *(const bf16x8*)(ap1 + kc * 32 + koff);
                bf16x8 hh1 = *(const bf16x8*)(ap1 + kc * 32 + koff + 16);
                a1[phase * 4 + kc] = __builtin_shufflevector(
                    lo1, hh1, 0,1,2,3,4,5,6,7,8,9,10,11,12,13,14,15);
            }
        }
        __syncthreads();
    }

    float bestv0[8], bestv1[8];
    int   besti0[8], besti1[8];
    #pragma unroll
    for (int r = 0; r < 8; ++r) {
        bestv0[r] = 3.4e38f; besti0[r] = 0;
        bestv1[r] = 3.4e38f; besti1[r] = 0;
    }

    // B fragment layout (16-bit B 32x16): lanes 0-15: K 0-7 & 16-23; lanes 16-31: K 8-15 & 24-31
    const int klo = hi * 8;
    #pragma unroll 1
    for (int jt = 0; jt < NCODE / 16; ++jt) {
        const int code = jt * 16 + m;
        const bf16_t* bp = cbb + code * DIM + klo;
        f32x8 acc0 = {0.f, 0.f, 0.f, 0.f, 0.f, 0.f, 0.f, 0.f};
        f32x8 acc1 = {0.f, 0.f, 0.f, 0.f, 0.f, 0.f, 0.f, 0.f};
        const float e2v = e2[code];
        #pragma unroll
        for (int kc = 0; kc < 8; ++kc) {
            bf16x8 lo = *(const bf16x8*)(bp + kc * 32);
            bf16x8 hh = *(const bf16x8*)(bp + kc * 32 + 16);
            bf16x16 bfrag = __builtin_shufflevector(
                lo, hh, 0,1,2,3,4,5,6,7,8,9,10,11,12,13,14,15);
            acc0 = __builtin_amdgcn_wmma_f32_16x16x32_bf16(
                false, a0[kc], false, bfrag, (short)0, acc0, false, false);
            acc1 = __builtin_amdgcn_wmma_f32_16x16x32_bf16(
                false, a1[kc], false, bfrag, (short)0, acc1, false, false);
        }
        #pragma unroll
        for (int r = 0; r < 8; ++r) {
            float s0 = fmaf(-2.0f, acc0[r], e2v);
            if (s0 < bestv0[r]) { bestv0[r] = s0; besti0[r] = code; }
            float s1 = fmaf(-2.0f, acc1[r], e2v);
            if (s1 < bestv1[r]) { bestv1[r] = s1; besti1[r] = code; }
        }
    }

    // butterfly min within each 16-lane half (C/D: lanes 0-15 rows 0-7, lanes 16-31 rows 8-15)
    #pragma unroll
    for (int off = 8; off >= 1; off >>= 1) {
        #pragma unroll
        for (int r = 0; r < 8; ++r) {
            float ov0 = __shfl_xor(bestv0[r], off, 32);
            int   oi0 = __shfl_xor(besti0[r], off, 32);
            if (ov0 < bestv0[r]) { bestv0[r] = ov0; besti0[r] = oi0; }
            float ov1 = __shfl_xor(bestv1[r], off, 32);
            int   oi1 = __shfl_xor(besti1[r], off, 32);
            if (ov1 < bestv1[r]) { bestv1[r] = ov1; besti1[r] = oi1; }
        }
    }
    if (m == 0) {
        const int base0 = n0 + wave * 32 + hi * 8;   // tile0 pixels
        const int base1 = base0 + 16;                // tile1 pixels
        #pragma unroll
        for (int r = 0; r < 8; ++r) {
            idxout[base0 + r] = besti0[r];
            idxout[base1 + r] = besti1[r];
        }
    }
}

// ---------------- Kernel 2: gather z_q (NCHW) + per-block squared-error ----------------
__global__ __launch_bounds__(256) void gather_kernel(
    const float* __restrict__ z, const float* __restrict__ cb,
    const int* __restrict__ idx, float* __restrict__ out,
    float* __restrict__ partial) {
    __shared__ float red[256];
    const int g  = blockIdx.x * 256 + threadIdx.x;   // NCHW flat index
    const int hw = g & 4095;
    const int c  = (g >> 12) & 255;
    const int b  = g >> 20;
    const int n  = (b << 12) | hw;                   // pixel id (b, hw)
    const int j  = idx[n];
    float e  = cb[j * DIM + c];                      // exact fp32 gather
    float zv = z[g];
    out[g] = e;                                      // straight-through value == z_q
    float d = e - zv;
    red[threadIdx.x] = d * d;
    __syncthreads();
    for (int s = 128; s > 0; s >>= 1) {
        if (threadIdx.x < s) red[threadIdx.x] += red[threadIdx.x + s];
        __syncthreads();
    }
    if (threadIdx.x == 0) partial[blockIdx.x] = red[0];
}

// ---------------- Kernel 3: deterministic final loss reduction ----------------
__global__ __launch_bounds__(256) void loss_kernel(
    const float* __restrict__ partial, float* __restrict__ loss) {
    __shared__ float red[256];
    float s = 0.f;
    for (int i = threadIdx.x; i < NOUT_Z / 256; i += 256) s += partial[i];
    red[threadIdx.x] = s;
    __syncthreads();
    for (int st = 128; st > 0; st >>= 1) {
        if (threadIdx.x < st) red[threadIdx.x] += red[threadIdx.x + st];
        __syncthreads();
    }
    // loss = (beta + 1) * mean(diff^2) = 1.25 * sum / (N*C)
    if (threadIdx.x == 0) loss[0] = 1.25f * red[0] / (float)NOUT_Z;
}

extern "C" void kernel_launch(void* const* d_in, const int* in_sizes, int n_in,
                              void* d_out, int out_size, void* d_ws, size_t ws_size,
                              hipStream_t stream) {
    const float* z  = (const float*)d_in[0];   // [8,256,64,64]
    const float* cb = (const float*)d_in[1];   // [1024,256]

    float* out   = (float*)d_out;              // z_q_out [8,256,64,64]
    float* loss  = out + NOUT_Z;               // scalar
    int*   idxo  = (int*)(out + NOUT_Z + 1);   // idx [8,64,64] as int32 bits

    // workspace: bf16 codebook (512KB) | e2 (4KB) | per-block partials (128KB)
    char* ws = (char*)d_ws;
    unsigned short* cbb = (unsigned short*)ws;
    float* e2      = (float*)(ws + (size_t)NCODE * DIM * 2);
    float* partial = (float*)(ws + (size_t)NCODE * DIM * 2 + NCODE * 4);

    prep_codebook<<<NCODE, 256, 0, stream>>>(cb, cbb, e2);
    argmin_kernel<<<NPIX / 128, 128, 0, stream>>>(z, (const bf16_t*)cbb, e2, idxo);
    gather_kernel<<<NOUT_Z / 256, 256, 0, stream>>>(z, cb, idxo, out, partial);
    loss_kernel<<<1, 256, 0, stream>>>(partial, loss);
}
